// FluxSingleTransformerBlock_61624190763527
// MI455X (gfx1250) — matrix operand order repaired and checked by reading, not compiled, not run on hardware
//
#include <hip/hip_runtime.h>
#include <hip/hip_bf16.h>
#include <math.h>
#include <stdint.h>

// ---------------------------------------------------------------------------
// Flux single transformer block for MI455X (gfx1250, wave32, WMMA + async LDS)
// B=1, S=2048, D=3072, H=24, DH=128, M=12288
// ---------------------------------------------------------------------------

typedef __attribute__((ext_vector_type(16))) _Float16 v16h;
typedef __attribute__((ext_vector_type(8)))  _Float16 v8h;
typedef __attribute__((ext_vector_type(8)))  float    v8f;

#define S_LEN 2048
#define DMODEL 3072
#define NHEAD 24
#define DHEAD 128
#define MDIM  12288
#define NQKVM (3 * DMODEL + MDIM)   // 21504
#define KOUT  (DMODEL + MDIM)       // 15360

// ---------------------------------------------------------------------------
// CDNA5 primitives
// ---------------------------------------------------------------------------
__device__ __forceinline__ v8f wmma_f16(v16h a, v16h b, v8f c) {
  return __builtin_amdgcn_wmma_f32_16x16x32_f16(false, a, false, b, (short)0, c,
                                                false, false);
}

// Async global->LDS copy of 16 bytes per lane (ASYNCcnt-tracked).
__device__ __forceinline__ void async_copy16(const void* g, void* lds) {
  asm volatile("global_load_async_to_lds_b128 %0, %1, off"
               :
               : "v"((uint32_t)(uintptr_t)lds), "v"((uint64_t)(uintptr_t)g)
               : "memory");
}

template <int N>
__device__ __forceinline__ void wait_async() {
  asm volatile("s_wait_asynccnt %0" : : "n"(N) : "memory");
}

// A fragment: 16x32 (MxK) f16, row-major, leading dim `ld` (halves).
// lane m=L%16, g=L/16; halves[i]=k=g*16+i  (one contiguous 32B run)
__device__ __forceinline__ v16h load_a_frag(const _Float16* base, int ld) {
  int lane = threadIdx.x & 31;
  int m = lane & 15, g = lane >> 4;
  const _Float16* p = base + (size_t)m * ld + g * 16;
  v8h lo = *(const v8h*)(p);
  v8h hi = *(const v8h*)(p + 8);
  v16h a;
#pragma unroll
  for (int i = 0; i < 8; ++i) { a[i] = lo[i]; a[8 + i] = hi[i]; }
  return a;
}

// B fragment: 32x16 (KxN); source N-major row-major (tile[n][k]).
// lane n=L%16, g=L/16; halves[i]=k=g*16+i  (one contiguous 32B run)
__device__ __forceinline__ v16h load_b_frag_nk(const _Float16* base, int ld) {
  int lane = threadIdx.x & 31;
  int n = lane & 15, g = lane >> 4;
  const _Float16* p = base + (size_t)n * ld + g * 16;
  v8h lo = *(const v8h*)(p);
  v8h hi = *(const v8h*)(p + 8);
  v16h b;
#pragma unroll
  for (int i = 0; i < 8; ++i) { b[i] = lo[i]; b[8 + i] = hi[i]; }
  return b;
}

// B fragment: 32x16 (KxN); source K-major row-major (tile[k][n]).
__device__ __forceinline__ v16h load_b_frag_kn(const _Float16* base, int ld) {
  int lane = threadIdx.x & 31;
  int n = lane & 15, g = lane >> 4;
  v16h b;
#pragma unroll
  for (int i = 0; i < 16; ++i) b[i] = base[(size_t)(g * 16 + i) * ld + n];
  return b;
}

__device__ __forceinline__ v8f v8f_zero() {
  v8f z;
#pragma unroll
  for (int i = 0; i < 8; ++i) z[i] = 0.0f;
  return z;
}

__device__ __forceinline__ float gelu_tanh(float x) {
  return 0.5f * x * (1.0f + tanhf(0.7978845608028654f * (x + 0.044715f * x * x * x)));
}

// ---------------------------------------------------------------------------
// Kernel 1: fp32 -> f16 repack (grid-stride)
// ---------------------------------------------------------------------------
__global__ void k_cvt_f16(const float* __restrict__ src, _Float16* __restrict__ dst,
                          long n) {
  long i = (long)blockIdx.x * blockDim.x + threadIdx.x;
  long stride = (long)gridDim.x * blockDim.x;
  for (; i < n; i += stride) dst[i] = (_Float16)src[i];
}

// ---------------------------------------------------------------------------
// Kernel 2: emb = silu(temb) @ norm_w.T + norm_b    ([9216]); wave per output
// ---------------------------------------------------------------------------
__global__ __launch_bounds__(256) void k_adaln_emb(const float* __restrict__ temb,
                                                   const float* __restrict__ w,
                                                   const float* __restrict__ b,
                                                   float* __restrict__ emb) {
  int wave = threadIdx.x >> 5, lane = threadIdx.x & 31;
  int n = blockIdx.x * 8 + wave;
  float s = 0.0f;
  const float* wr = w + (size_t)n * DMODEL;
  for (int k = lane; k < DMODEL; k += 32) {
    float t = temb[k];
    s += (t / (1.0f + __expf(-t))) * wr[k];
  }
#pragma unroll
  for (int off = 16; off >= 1; off >>= 1) s += __shfl_xor(s, off);
  if (lane == 0) emb[n] = s + b[n];
}

// ---------------------------------------------------------------------------
// Kernel 3: LayerNorm + (1+scale)*x + shift -> nx (f16); block per token
// ---------------------------------------------------------------------------
__global__ __launch_bounds__(256) void k_adaln_norm(const float* __restrict__ x,
                                                    const float* __restrict__ emb,
                                                    _Float16* __restrict__ nx) {
  int s = blockIdx.x;
  const float* xr = x + (size_t)s * DMODEL;
  int lane = threadIdx.x & 31, wave = threadIdx.x >> 5;
  float sum = 0.0f, sq = 0.0f;
  for (int d = threadIdx.x; d < DMODEL; d += 256) {
    float v = xr[d];
    sum += v;
    sq += v * v;
  }
#pragma unroll
  for (int off = 16; off >= 1; off >>= 1) {
    sum += __shfl_xor(sum, off);
    sq += __shfl_xor(sq, off);
  }
  __shared__ float ssum[8], ssq[8];
  if (lane == 0) { ssum[wave] = sum; ssq[wave] = sq; }
  __syncthreads();
  float ts = 0.0f, tq = 0.0f;
#pragma unroll
  for (int i = 0; i < 8; ++i) { ts += ssum[i]; tq += ssq[i]; }
  float mu = ts * (1.0f / DMODEL);
  float var = tq * (1.0f / DMODEL) - mu * mu;
  float rstd = rsqrtf(var + 1e-6f);
  for (int d = threadIdx.x; d < DMODEL; d += 256) {
    float v = (xr[d] - mu) * rstd;
    v = v * (1.0f + emb[DMODEL + d]) + emb[d];
    nx[(size_t)s * DMODEL + d] = (_Float16)v;
  }
}

// ---------------------------------------------------------------------------
// Async tile issue for GEMMs: A tile 64x64, B tile 256x64 (halves), 256 thr.
// 10 async instructions per thread per tile.
// ---------------------------------------------------------------------------
__device__ __forceinline__ void gemm_issue_tile(const _Float16* Asrc, int lda,
                                                int kkA, const _Float16* Bw,
                                                int ldb, int k0, int mblk,
                                                int nblk, _Float16* AsBuf,
                                                _Float16* BsBuf, int t) {
  int arow = t >> 2, acb = (t & 3) * 16;
#pragma unroll
  for (int op = 0; op < 2; ++op)
    async_copy16(&Asrc[(size_t)(mblk + arow) * lda + kkA + acb + op * 8],
                 &AsBuf[arow * 64 + acb + op * 8]);
#pragma unroll
  for (int op = 0; op < 8; ++op)
    async_copy16(&Bw[(size_t)(nblk + t) * ldb + k0 + op * 8],
                 &BsBuf[t * 64 + op * 8]);
}

// ---------------------------------------------------------------------------
// Kernel 4: fused QKV + MLP GEMM:  C[2048, 21504] = nx @ Wcat^T  (WMMA)
// 64(M) x 256(N) workgroup tile, 8 waves (2x4), wave tile 32x64,
// K-step 64, async double-buffered LDS.
// ---------------------------------------------------------------------------
__global__ __launch_bounds__(256) void k_gemm_qkvm(
    const _Float16* __restrict__ A,    // [2048][3072]
    const _Float16* __restrict__ Bw,   // [21504][3072]
    const float* __restrict__ qb, const float* __restrict__ kb,
    const float* __restrict__ vb, const float* __restrict__ mb,
    _Float16* __restrict__ qkv,        // [2048][9216]
    _Float16* __restrict__ mlp)        // [2048][12288]
{
  const int K = DMODEL, NT = K / 64;
  int t = threadIdx.x;
  int wave = t >> 5, lane = t & 31;
  int wm = wave >> 2, wn = wave & 3;
  int mblk = blockIdx.y * 64, nblk = blockIdx.x * 256;

  __shared__ alignas(16) _Float16 As[2][64 * 64];
  __shared__ alignas(16) _Float16 Bs[2][256 * 64];

  v8f acc[2][4];
#pragma unroll
  for (int i = 0; i < 2; ++i)
#pragma unroll
    for (int j = 0; j < 4; ++j) acc[i][j] = v8f_zero();

  gemm_issue_tile(A, K, 0, Bw, K, 0, mblk, nblk, As[0], Bs[0], t);

  for (int kt = 0; kt < NT; ++kt) {
    int cur = kt & 1;
    if (kt + 1 < NT) {
      gemm_issue_tile(A, K, (kt + 1) * 64, Bw, K, (kt + 1) * 64, mblk, nblk,
                      As[cur ^ 1], Bs[cur ^ 1], t);
      wait_async<10>();
    } else {
      wait_async<0>();
    }
    __syncthreads();   // tile kt visible to all waves
#pragma unroll
    for (int ks = 0; ks < 2; ++ks) {
      v16h a0 = load_a_frag(&As[cur][(wm * 32) * 64 + ks * 32], 64);
      v16h a1 = load_a_frag(&As[cur][(wm * 32 + 16) * 64 + ks * 32], 64);
#pragma unroll
      for (int nf = 0; nf < 4; ++nf) {
        v16h b = load_b_frag_nk(&Bs[cur][(wn * 64 + nf * 16) * 64 + ks * 32], 64);
        acc[0][nf] = wmma_f16(a0, b, acc[0][nf]);
        acc[1][nf] = wmma_f16(a1, b, acc[1][nf]);
      }
    }
    __syncthreads();   // all waves done reading buf before it is refilled
  }

  int lh = lane >> 4, ln = lane & 15;
#pragma unroll
  for (int mi = 0; mi < 2; ++mi)
#pragma unroll
    for (int nf = 0; nf < 4; ++nf)
#pragma unroll
      for (int j = 0; j < 8; ++j) {
        int row = mblk + wm * 32 + mi * 16 + j + 8 * lh;
        int col = nblk + wn * 64 + nf * 16 + ln;
        float v = acc[mi][nf][j];
        float bias = (col < DMODEL)       ? qb[col]
                     : (col < 2 * DMODEL) ? kb[col - DMODEL]
                     : (col < 3 * DMODEL) ? vb[col - 2 * DMODEL]
                                          : mb[col - 3 * DMODEL];
        v += bias;
        if (col < 3 * DMODEL) {
          qkv[(size_t)row * (3 * DMODEL) + col] = (_Float16)v;
        } else {
          mlp[(size_t)row * MDIM + (col - 3 * DMODEL)] = (_Float16)gelu_tanh(v);
        }
      }
}

// ---------------------------------------------------------------------------
// Kernel 5: per-(token,head) RMSNorm + RoPE; layout change to [h][s][128] f16
// ---------------------------------------------------------------------------
__global__ __launch_bounds__(256) void k_rms_rope(
    const _Float16* __restrict__ qkv, const float* __restrict__ cosb,
    const float* __restrict__ sinb, const float* __restrict__ qrw,
    const float* __restrict__ krw, _Float16* __restrict__ qh,
    _Float16* __restrict__ kh, _Float16* __restrict__ vh) {
  int gw = blockIdx.x * 8 + (threadIdx.x >> 5);
  int lane = threadIdx.x & 31;
  int s = gw / NHEAD, h = gw % NHEAD;
  int i0 = lane * 4;
  size_t hso = ((size_t)h * S_LEN + s) * DHEAD + i0;
  const float* cp = cosb + (size_t)s * DHEAD + i0;
  const float* sp = sinb + (size_t)s * DHEAD + i0;

#pragma unroll
  for (int which = 0; which < 2; ++which) {   // 0: q, 1: k
    const _Float16* p =
        qkv + (size_t)s * (3 * DMODEL) + which * DMODEL + h * DHEAD + i0;
    const float* rw = which ? krw : qrw;
    float x0 = (float)p[0], x1 = (float)p[1], x2 = (float)p[2], x3 = (float)p[3];
    float ss = x0 * x0 + x1 * x1 + x2 * x2 + x3 * x3;
#pragma unroll
    for (int off = 16; off >= 1; off >>= 1) ss += __shfl_xor(ss, off);
    float r = rsqrtf(ss * (1.0f / DHEAD) + 1e-6f);
    x0 *= r * rw[i0 + 0]; x1 *= r * rw[i0 + 1];
    x2 *= r * rw[i0 + 2]; x3 *= r * rw[i0 + 3];
    float y0 = x0 * cp[0] - x1 * sp[0];
    float y1 = x1 * cp[1] + x0 * sp[1];
    float y2 = x2 * cp[2] - x3 * sp[2];
    float y3 = x3 * cp[3] + x2 * sp[3];
    _Float16* dst = (which ? kh : qh) + hso;
    dst[0] = (_Float16)y0; dst[1] = (_Float16)y1;
    dst[2] = (_Float16)y2; dst[3] = (_Float16)y3;
  }
  const _Float16* pv = qkv + (size_t)s * (3 * DMODEL) + 2 * DMODEL + h * DHEAD + i0;
  _Float16* dv = vh + hso;
#pragma unroll
  for (int i = 0; i < 4; ++i) dv[i] = pv[i];
}

// ---------------------------------------------------------------------------
// Kernel 6: flash attention with WMMA, async double-buffered K/V tiles
// grid (32 q-tiles, 24 heads), 128 threads (4 waves), wave owns 16 q rows
// ---------------------------------------------------------------------------
__global__ __launch_bounds__(128) void k_attn(const _Float16* __restrict__ qh,
                                              const _Float16* __restrict__ kh,
                                              const _Float16* __restrict__ vh,
                                              _Float16* __restrict__ attn_out) {
  int h = blockIdx.y;
  int wave = threadIdx.x >> 5, lane = threadIdx.x & 31;
  int qrow0 = blockIdx.x * 64 + wave * 16;
  const _Float16* Q = qh + (size_t)h * S_LEN * DHEAD;
  const _Float16* Kp = kh + (size_t)h * S_LEN * DHEAD;
  const _Float16* Vp = vh + (size_t)h * S_LEN * DHEAD;

  __shared__ alignas(16) _Float16 Kt[2][32 * DHEAD];
  __shared__ alignas(16) _Float16 Vt[2][32 * DHEAD];
  __shared__ alignas(16) _Float16 Pb[4][16 * 32];

  int r = threadIdx.x >> 2, c = (threadIdx.x & 3) * 32;
  // issue tile 0 (8 async ops per thread)
#pragma unroll
  for (int j = 0; j < 4; ++j) {
    async_copy16(&Kp[(size_t)r * DHEAD + c + j * 8], &Kt[0][r * DHEAD + c + j * 8]);
    async_copy16(&Vp[(size_t)r * DHEAD + c + j * 8], &Vt[0][r * DHEAD + c + j * 8]);
  }

  v16h qa[4];
#pragma unroll
  for (int kc = 0; kc < 4; ++kc)
    qa[kc] = load_a_frag(Q + (size_t)qrow0 * DHEAD + kc * 32, DHEAD);

  v8f oacc[8];
#pragma unroll
  for (int i = 0; i < 8; ++i) oacc[i] = v8f_zero();
  float mrun[8], lrun[8];
#pragma unroll
  for (int j = 0; j < 8; ++j) { mrun[j] = -1e30f; lrun[j] = 0.0f; }

  const float scale = 0.08838834764831845f;  // 1/sqrt(128)
  const int NT = S_LEN / 32;

  for (int kt = 0; kt < NT; ++kt) {
    int cur = kt & 1;
    if (kt + 1 < NT) {
      int nb = cur ^ 1, krow = (kt + 1) * 32;
#pragma unroll
      for (int j = 0; j < 4; ++j) {
        async_copy16(&Kp[(size_t)(krow + r) * DHEAD + c + j * 8],
                     &Kt[nb][r * DHEAD + c + j * 8]);
        async_copy16(&Vp[(size_t)(krow + r) * DHEAD + c + j * 8],
                     &Vt[nb][r * DHEAD + c + j * 8]);
      }
      wait_async<8>();
    } else {
      wait_async<0>();
    }
    __syncthreads();

    // scores: 16 q-rows x 32 keys
    v8f sc[2];
#pragma unroll
    for (int nf = 0; nf < 2; ++nf) {
      sc[nf] = v8f_zero();
#pragma unroll
      for (int kc = 0; kc < 4; ++kc) {
        v16h b = load_b_frag_nk(&Kt[cur][(nf * 16) * DHEAD + kc * 32], DHEAD);
        sc[nf] = wmma_f16(qa[kc], b, sc[nf]);
      }
    }

    // online softmax; C layout: VGPR j = row j (lanes 0-15) / j+8 (lanes 16-31)
    float corr[8];
#pragma unroll
    for (int j = 0; j < 8; ++j) {
      float s0 = sc[0][j] * scale;
      float s1 = sc[1][j] * scale;
      float mx = fmaxf(s0, s1);
#pragma unroll
      for (int off = 8; off >= 1; off >>= 1) mx = fmaxf(mx, __shfl_xor(mx, off));
      float mnew = fmaxf(mrun[j], mx);
      corr[j] = __expf(mrun[j] - mnew);
      mrun[j] = mnew;
      float p0 = __expf(s0 - mnew);
      float p1 = __expf(s1 - mnew);
      float ls = p0 + p1;
#pragma unroll
      for (int off = 8; off >= 1; off >>= 1) ls += __shfl_xor(ls, off);
      lrun[j] = lrun[j] * corr[j] + ls;
      sc[0][j] = p0;
      sc[1][j] = p1;
    }
#pragma unroll
    for (int df = 0; df < 8; ++df)
#pragma unroll
      for (int j = 0; j < 8; ++j) oacc[df][j] *= corr[j];

    // transpose P through per-wave LDS: C layout -> A layout
#pragma unroll
    for (int nf = 0; nf < 2; ++nf)
#pragma unroll
      for (int j = 0; j < 8; ++j) {
        int m = j + 8 * (lane >> 4);
        int n = nf * 16 + (lane & 15);
        Pb[wave][m * 32 + n] = (_Float16)sc[nf][j];
      }
    asm volatile("s_wait_dscnt 0" ::: "memory");   // per-wave LDS RAW
    v16h pa = load_a_frag(&Pb[wave][0], 32);

    // O += P(16x32) @ V(32x128)
#pragma unroll
    for (int df = 0; df < 8; ++df) {
      v16h b = load_b_frag_kn(&Vt[cur][df * 16], DHEAD);
      oacc[df] = wmma_f16(pa, b, oacc[df]);
    }
    __syncthreads();   // all waves done with buf before refill
  }

#pragma unroll
  for (int df = 0; df < 8; ++df)
#pragma unroll
    for (int j = 0; j < 8; ++j) {
      int row = qrow0 + j + 8 * (lane >> 4);
      int col = h * DHEAD + df * 16 + (lane & 15);
      attn_out[(size_t)row * DMODEL + col] = (_Float16)(oacc[df][j] / lrun[j]);
    }
}

// ---------------------------------------------------------------------------
// Kernel 7: out projection: out = hidden + gate * ([attn | mlp] @ out_w^T + b)
// ---------------------------------------------------------------------------
__global__ __launch_bounds__(256) void k_gemm_out(
    const _Float16* __restrict__ Aattn,  // [2048][3072]
    const _Float16* __restrict__ Amlp,   // [2048][12288]
    const _Float16* __restrict__ Bw,     // [3072][15360]
    const float* __restrict__ out_b, const float* __restrict__ emb,
    const float* __restrict__ hidden, float* __restrict__ out) {
  const int NT = KOUT / 64;
  int t = threadIdx.x;
  int wave = t >> 5, lane = t & 31;
  int wm = wave >> 2, wn = wave & 3;
  int mblk = blockIdx.y * 64, nblk = blockIdx.x * 256;

  __shared__ alignas(16) _Float16 As[2][64 * 64];
  __shared__ alignas(16) _Float16 Bs[2][256 * 64];

  v8f acc[2][4];
#pragma unroll
  for (int i = 0; i < 2; ++i)
#pragma unroll
    for (int j = 0; j < 4; ++j) acc[i][j] = v8f_zero();

  gemm_issue_tile(Aattn, DMODEL, 0, Bw, KOUT, 0, mblk, nblk, As[0], Bs[0], t);

  for (int kt = 0; kt < NT; ++kt) {
    int cur = kt & 1;
    if (kt + 1 < NT) {
      int k0 = (kt + 1) * 64;
      const _Float16* asrc;
      int lda, kk;
      if (k0 < DMODEL) { asrc = Aattn; lda = DMODEL; kk = k0; }
      else             { asrc = Amlp;  lda = MDIM;   kk = k0 - DMODEL; }
      gemm_issue_tile(asrc, lda, kk, Bw, KOUT, k0, mblk, nblk, As[cur ^ 1],
                      Bs[cur ^ 1], t);
      wait_async<10>();
    } else {
      wait_async<0>();
    }
    __syncthreads();
#pragma unroll
    for (int ks = 0; ks < 2; ++ks) {
      v16h a0 = load_a_frag(&As[cur][(wm * 32) * 64 + ks * 32], 64);
      v16h a1 = load_a_frag(&As[cur][(wm * 32 + 16) * 64 + ks * 32], 64);
#pragma unroll
      for (int nf = 0; nf < 4; ++nf) {
        v16h b = load_b_frag_nk(&Bs[cur][(wn * 64 + nf * 16) * 64 + ks * 32], 64);
        acc[0][nf] = wmma_f16(a0, b, acc[0][nf]);
        acc[1][nf] = wmma_f16(a1, b, acc[1][nf]);
      }
    }
    __syncthreads();
  }

  int lh = lane >> 4, ln = lane & 15;
#pragma unroll
  for (int mi = 0; mi < 2; ++mi)
#pragma unroll
    for (int nf = 0; nf < 4; ++nf)
#pragma unroll
      for (int j = 0; j < 8; ++j) {
        int row = mblk + wm * 32 + mi * 16 + j + 8 * lh;
        int col = nblk + wn * 64 + nf * 16 + ln;
        float v = acc[mi][nf][j] + out_b[col];
        float gate = emb[2 * DMODEL + col];
        size_t idx = (size_t)row * DMODEL + col;
        out[idx] = hidden[idx] + gate * v;
      }
}

// ---------------------------------------------------------------------------
// Host launcher
// ---------------------------------------------------------------------------
extern "C" void kernel_launch(void* const* d_in, const int* in_sizes, int n_in,
                              void* d_out, int out_size, void* d_ws, size_t ws_size,
                              hipStream_t stream) {
  const float* hidden   = (const float*)d_in[0];
  const float* temb     = (const float*)d_in[1];
  const float* rope_cos = (const float*)d_in[2];
  const float* rope_sin = (const float*)d_in[3];
  const float* norm_w   = (const float*)d_in[4];
  const float* norm_b   = (const float*)d_in[5];
  const float* qw       = (const float*)d_in[6];
  const float* qb       = (const float*)d_in[7];
  const float* kw       = (const float*)d_in[8];
  const float* kb       = (const float*)d_in[9];
  const float* vw       = (const float*)d_in[10];
  const float* vb       = (const float*)d_in[11];
  const float* q_rms_w  = (const float*)d_in[12];
  const float* k_rms_w  = (const float*)d_in[13];
  const float* mlp_w    = (const float*)d_in[14];
  const float* mlp_b    = (const float*)d_in[15];
  const float* out_w    = (const float*)d_in[16];
  const float* out_b    = (const float*)d_in[17];

  char* ws = (char*)d_ws;
  size_t off = 0;
  auto alloc = [&](size_t bytes) {
    void* p = ws + off;
    off = (off + bytes + 255) & ~(size_t)255;
    return p;
  };
  _Float16* wqkvm = (_Float16*)alloc((size_t)NQKVM * DMODEL * 2);
  _Float16* wout  = (_Float16*)alloc((size_t)DMODEL * KOUT * 2);
  float*    emb   = (float*)   alloc((size_t)3 * DMODEL * 4);
  _Float16* nx    = (_Float16*)alloc((size_t)S_LEN * DMODEL * 2);
  _Float16* qkv   = (_Float16*)alloc((size_t)S_LEN * 3 * DMODEL * 2);
  _Float16* mlp   = (_Float16*)alloc((size_t)S_LEN * MDIM * 2);
  _Float16* qhb   = (_Float16*)alloc((size_t)S_LEN * DMODEL * 2);
  _Float16* khb   = (_Float16*)alloc((size_t)S_LEN * DMODEL * 2);
  _Float16* vhb   = (_Float16*)alloc((size_t)S_LEN * DMODEL * 2);
  _Float16* attn  = (_Float16*)alloc((size_t)S_LEN * DMODEL * 2);

  long dd = (long)DMODEL * DMODEL;
  k_cvt_f16<<<4096, 256, 0, stream>>>(qw, wqkvm, dd);
  k_cvt_f16<<<4096, 256, 0, stream>>>(kw, wqkvm + dd, dd);
  k_cvt_f16<<<4096, 256, 0, stream>>>(vw, wqkvm + 2 * dd, dd);
  k_cvt_f16<<<8192, 256, 0, stream>>>(mlp_w, wqkvm + 3 * dd, (long)MDIM * DMODEL);
  k_cvt_f16<<<8192, 256, 0, stream>>>(out_w, wout, (long)DMODEL * KOUT);

  k_adaln_emb<<<(3 * DMODEL) / 8, 256, 0, stream>>>(temb, norm_w, norm_b, emb);
  k_adaln_norm<<<S_LEN, 256, 0, stream>>>(hidden, emb, nx);

  k_gemm_qkvm<<<dim3(NQKVM / 256, S_LEN / 64), 256, 0, stream>>>(
      nx, wqkvm, qb, kb, vb, mlp_b, qkv, mlp);

  k_rms_rope<<<(S_LEN * NHEAD) / 8, 256, 0, stream>>>(qkv, rope_cos, rope_sin,
                                                      q_rms_w, k_rms_w, qhb, khb,
                                                      vhb);

  k_attn<<<dim3(S_LEN / 64, NHEAD), 128, 0, stream>>>(qhb, khb, vhb, attn);

  k_gemm_out<<<dim3(DMODEL / 256, S_LEN / 64), 256, 0, stream>>>(
      attn, mlp, wout, out_b, emb, hidden, (float*)d_out);
}
